// EGCN_6786048328676
// MI455X (gfx1250) — matrix operand order repaired and checked by reading, not compiled, not run on hardware
//
#include <hip/hip_runtime.h>
#include <cstdint>

// N = 8192 nodes, D = 128 features.
#define NN 8192
#define DD 128

typedef __bf16 bf16_t;
typedef __attribute__((ext_vector_type(16))) __bf16 v16bf;
typedef __attribute__((ext_vector_type(8)))  float  v8f;

// Issue one async 16-byte global->LDS copy (per active lane). Tracked by ASYNCcnt.
#define ASYNC_CP_B128(ldsoff, gptr)                                          \
    asm volatile("global_load_async_to_lds_b128 %0, %1, off"                 \
                 :: "v"(ldsoff), "v"(gptr) : "memory")

#define WAIT_ASYNC_LE4() asm volatile("s_wait_asynccnt 0x4" ::: "memory")
#define WAIT_ASYNC_LE0() asm volatile("s_wait_asynccnt 0x0" ::: "memory")

// ---------------------------------------------------------------------------
// Convert embeddings [N][D] f32 -> transposed bf16 [D][N] (column-major for
// WMMA B-operand fragments: 16 contiguous K values per lane).
// ---------------------------------------------------------------------------
__global__ void k_cvt_xT(const float* __restrict__ emb, bf16_t* __restrict__ xbT) {
    int t = blockIdx.x * blockDim.x + threadIdx.x;   // N*D threads
    int i = t & (NN - 1);       // node index (fast -> coalesced bf16 writes)
    int c = t >> 13;            // feature index
    xbT[(size_t)c * NN + i] = (bf16_t)emb[(size_t)i * DD + c];
}

// Convert weight [D][D] f32 -> transposed bf16 [D(out)][D(in)].
__global__ void k_cvt_wT(const float* __restrict__ W, bf16_t* __restrict__ wbT) {
    int t = blockIdx.x * blockDim.x + threadIdx.x;   // D*D threads
    int k = t & (DD - 1);
    int c = t >> 7;
    wbT[c * DD + k] = (bf16_t)W[k * DD + c];
}

// ---------------------------------------------------------------------------
// Big GEMM: y[N][D] = A[N][N] (f32 streamed via async-LDS, cvt->bf16) @ xbT^T.
// Block = 128 threads (4 waves). Each wave owns a private 16-row strip and a
// private double-buffered 16x32 f32 LDS tile filled with
// GLOBAL_LOAD_ASYNC_TO_LDS_B128 (coalesced 128B row lines), prefetched one
// k-step ahead (ASYNCcnt). Full K per wave => deg rowsum fused (layer 0),
// deterministic, no atomics, no barriers.
// ---------------------------------------------------------------------------
template <bool COMPUTE_DEG>
__global__ __launch_bounds__(128)
void k_gemm(const float* __restrict__ A, const bf16_t* __restrict__ xbT,
            float* __restrict__ y, float* __restrict__ deg) {
    constexpr int RS  = 36;          // padded LDS row stride (floats) -> 16 distinct banks
    constexpr int BUF = 16 * RS;     // floats per buffer (2304 B)
    __shared__ float sA[4 * 2 * BUF];   // 4 waves x 2 buffers = 18 KB

    const int lane  = threadIdx.x & 31;
    const int wave  = threadIdx.x >> 5;
    const int mlane = lane & 15;
    const int hi    = lane >> 4;       // lane half (0/1)
    const int hi8   = hi << 3;
    const int rowStrip = blockIdx.x * 64 + wave * 16;

    // Async-copy lane mapping: tile = 16 rows x 32 floats; chunk = 16 B.
    // chunk c = i*32 + lane ; row = c>>3 ; col = (c&7)*4  (8 chunks per row ->
    // lanes 0..7 cover one full 128B row line: perfectly coalesced).
    const float* gsrc[4];
    unsigned     ldst[4];
#pragma unroll
    for (int i = 0; i < 4; ++i) {
        int c   = i * 32 + lane;
        int r   = c >> 3;
        int col = (c & 7) * 4;
        gsrc[i] = A + (size_t)(rowStrip + r) * NN + col;
        ldst[i] = (unsigned)(uintptr_t)&sA[(wave * 2) * BUF + r * RS + col];
    }
    const unsigned BUFB = (unsigned)BUF * 4u;   // buffer stride in bytes

    // Prologue: fill buffer 0 with the kk=0 tile.
#pragma unroll
    for (int i = 0; i < 4; ++i) ASYNC_CP_B128(ldst[i], gsrc[i]);

    const bf16_t* __restrict__ xcol = xbT + (size_t)mlane * NN + (hi << 4);
    const float*  __restrict__ bufw = &sA[(wave * 2) * BUF + mlane * RS];

    v8f acc[8];
#pragma unroll
    for (int n = 0; n < 8; ++n) acc[n] = (v8f){0.f,0.f,0.f,0.f,0.f,0.f,0.f,0.f};

    float degp = 0.f;
    int ph = 0;

#pragma unroll 1
    for (int kk = 0; kk < NN; kk += 32) {
        if (kk + 32 < NN) {
            // Prefetch next tile into the other buffer.
            unsigned bo = (unsigned)(ph ^ 1) * BUFB;
#pragma unroll
            for (int i = 0; i < 4; ++i)
                ASYNC_CP_B128(ldst[i] + bo, gsrc[i] + kk + 32);
            WAIT_ASYNC_LE4();   // in-order: <=4 outstanding => current tile landed
        } else {
            WAIT_ASYNC_LE0();
        }

        // A-matrix 16x32 bf16 fragment from LDS: lane half hi covers K run
        // [hi*16 .. +15].
        const float* bp = bufw + ph * BUF;
        float4 a0 = *(const float4*)(bp + (hi << 4));
        float4 a1 = *(const float4*)(bp + (hi << 4) + 4);
        float4 a2 = *(const float4*)(bp + (hi << 4) + 8);
        float4 a3 = *(const float4*)(bp + (hi << 4) + 12);

        v16bf af;
        af[0]=(bf16_t)a0.x;  af[1]=(bf16_t)a0.y;  af[2]=(bf16_t)a0.z;  af[3]=(bf16_t)a0.w;
        af[4]=(bf16_t)a1.x;  af[5]=(bf16_t)a1.y;  af[6]=(bf16_t)a1.z;  af[7]=(bf16_t)a1.w;
        af[8]=(bf16_t)a2.x;  af[9]=(bf16_t)a2.y;  af[10]=(bf16_t)a2.z; af[11]=(bf16_t)a2.w;
        af[12]=(bf16_t)a3.x; af[13]=(bf16_t)a3.y; af[14]=(bf16_t)a3.z; af[15]=(bf16_t)a3.w;

        if (COMPUTE_DEG) {
            degp += (a0.x + a0.y + a0.z + a0.w) + (a1.x + a1.y + a1.z + a1.w)
                  + (a2.x + a2.y + a2.z + a2.w) + (a3.x + a3.y + a3.z + a3.w);
        }

#pragma unroll
        for (int n = 0; n < 8; ++n) {
            // B-matrix 32x16 bf16: lane n-col = n*16+mlane, 16 contiguous K.
            v16bf bfr = *(const v16bf*)(xcol + ((size_t)n << 4) * NN + kk);
            acc[n] = __builtin_amdgcn_wmma_f32_16x16x32_bf16(
                false, af, false, bfr, (short)0, acc[n], false, false);
        }
        ph ^= 1;
    }

    if (COMPUTE_DEG) {
        // lane m and m+16 hold complementary K halves of the same row.
        float other = __shfl_xor(degp, 16, 32);
        if (hi == 0) deg[rowStrip + mlane] = degp + other;
    }

    // C/D layout: VGPR r -> row rowStrip + hi*8 + r, col n*16 + mlane.
    float* __restrict__ yrow = y + (size_t)(rowStrip + hi8) * DD + mlane;
#pragma unroll
    for (int n = 0; n < 8; ++n)
#pragma unroll
        for (int r = 0; r < 8; ++r)
            yrow[(size_t)r * DD + (n << 4)] = acc[n][r];
}

// ---------------------------------------------------------------------------
// Epilogue GEMM: h = relu((y / deg) @ W).  M=8192, K=128, N=128.
// Writes f32 row-major (final output) and/or bf16 transposed (next layer's B).
// ---------------------------------------------------------------------------
__global__ __launch_bounds__(128)
void k_norm_w(const float* __restrict__ y, const float* __restrict__ deg,
              const bf16_t* __restrict__ wbT,
              float* __restrict__ outF, bf16_t* __restrict__ outT) {
    const int lane  = threadIdx.x & 31;
    const int wave  = threadIdx.x >> 5;
    const int mlane = lane & 15;
    const int hi    = lane >> 4;
    const int hi8   = hi << 3;
    const int rowStrip = blockIdx.x * 64 + wave * 16;
    const int row = rowStrip + mlane;

    const float rdeg = 1.0f / deg[row];
    const float*  __restrict__ Yrow = y   + (size_t)row * DD;
    const bf16_t* __restrict__ wcol = wbT + mlane * DD + (hi << 4);

    v8f acc[8];
#pragma unroll
    for (int n = 0; n < 8; ++n) acc[n] = (v8f){0.f,0.f,0.f,0.f,0.f,0.f,0.f,0.f};

#pragma unroll
    for (int kk = 0; kk < DD; kk += 32) {
        const float4* p0 = (const float4*)(Yrow + kk + (hi << 4));
        const float4* p1 = (const float4*)(Yrow + kk + (hi << 4) + 8);
        float4 a0 = p0[0], a1 = p0[1];
        float4 a2 = p1[0], a3 = p1[1];

        v16bf af;
        af[0]=(bf16_t)(a0.x*rdeg);  af[1]=(bf16_t)(a0.y*rdeg);
        af[2]=(bf16_t)(a0.z*rdeg);  af[3]=(bf16_t)(a0.w*rdeg);
        af[4]=(bf16_t)(a1.x*rdeg);  af[5]=(bf16_t)(a1.y*rdeg);
        af[6]=(bf16_t)(a1.z*rdeg);  af[7]=(bf16_t)(a1.w*rdeg);
        af[8]=(bf16_t)(a2.x*rdeg);  af[9]=(bf16_t)(a2.y*rdeg);
        af[10]=(bf16_t)(a2.z*rdeg); af[11]=(bf16_t)(a2.w*rdeg);
        af[12]=(bf16_t)(a3.x*rdeg); af[13]=(bf16_t)(a3.y*rdeg);
        af[14]=(bf16_t)(a3.z*rdeg); af[15]=(bf16_t)(a3.w*rdeg);

#pragma unroll
        for (int n = 0; n < 8; ++n) {
            v16bf bfr = *(const v16bf*)(wcol + (n << 4) * DD + kk);
            acc[n] = __builtin_amdgcn_wmma_f32_16x16x32_bf16(
                false, af, false, bfr, (short)0, acc[n], false, false);
        }
    }

#pragma unroll
    for (int n = 0; n < 8; ++n)
#pragma unroll
        for (int r = 0; r < 8; ++r) {
            float v = fmaxf(acc[n][r], 0.0f);
            int ro = rowStrip + hi8 + r;
            int co = (n << 4) + mlane;
            if (outF) outF[(size_t)ro * DD + co] = v;
            if (outT) outT[(size_t)co * NN + ro] = (bf16_t)v;
        }
}

// ---------------------------------------------------------------------------
// Host launch sequence (graph-capture safe: only kernel launches on stream).
// ---------------------------------------------------------------------------
extern "C" void kernel_launch(void* const* d_in, const int* in_sizes, int n_in,
                              void* d_out, int out_size, void* d_ws, size_t ws_size,
                              hipStream_t stream) {
    const float* emb = (const float*)d_in[0];   // [N][D]
    const float* A   = (const float*)d_in[1];   // [N][N]
    const float* W0  = (const float*)d_in[2];   // [D][D]
    const float* W1  = (const float*)d_in[3];   // [D][D]
    float* out = (float*)d_out;                 // [N][D]

    char* ws = (char*)d_ws;
    const size_t XBT_BYTES = (size_t)DD * NN * sizeof(bf16_t);  // 2 MB
    const size_t Y_BYTES   = (size_t)NN * DD * sizeof(float);   // 4 MB
    const size_t DEG_BYTES = (size_t)NN * sizeof(float);        // 32 KB
    bf16_t* xbT = (bf16_t*)ws;
    float*  y   = (float*)(ws + XBT_BYTES);
    float*  deg = (float*)(ws + XBT_BYTES + Y_BYTES);
    bf16_t* wbT = (bf16_t*)(ws + XBT_BYTES + Y_BYTES + DEG_BYTES);
    (void)in_sizes; (void)n_in; (void)out_size; (void)ws_size;

    // layer 0
    k_cvt_xT<<<(NN * DD) / 256, 256, 0, stream>>>(emb, xbT);
    k_cvt_wT<<<(DD * DD) / 256, 256, 0, stream>>>(W0, wbT);
    k_gemm<true><<<NN / 64, 128, 0, stream>>>(A, xbT, y, deg);
    k_norm_w<<<NN / 64, 128, 0, stream>>>(y, deg, wbT, nullptr, xbT);

    // layer 1
    k_cvt_wT<<<(DD * DD) / 256, 256, 0, stream>>>(W1, wbT);
    k_gemm<false><<<NN / 64, 128, 0, stream>>>(A, xbT, y, nullptr);
    k_norm_w<<<NN / 64, 128, 0, stream>>>(y, deg, wbT, out, nullptr);
}
